// SelfAttention_50500225466674
// MI455X (gfx1250) — matrix operand order repaired and checked by reading, not compiled, not run on hardware
//
#include <hip/hip_runtime.h>

// Problem constants (from reference): x[N][D], Wq/Wk[H][Z][D], H*Z == 2048
#define N_TOK 2048
#define D_DIM 128
#define H_HEADS 32
#define Z_DIM 64
#define HZ_DIM 2048   // H*Z

// ---- WMMA operand types (CDNA5 wave32, 16x16x32 bf16) ----
typedef __bf16       v16bf __attribute__((ext_vector_type(16)));
typedef float        v8f   __attribute__((ext_vector_type(8)));
typedef unsigned int u32x4 __attribute__((ext_vector_type(4)));

union V16U { v16bf v; u32x4 q[2]; };

#if __has_builtin(__builtin_amdgcn_exp2f)
#define EXP2F(x) __builtin_amdgcn_exp2f(x)
#else
#define EXP2F(x) exp2f(x)
#endif
#if __has_builtin(__builtin_amdgcn_logf)
#define LOG2F(x) __builtin_amdgcn_logf(x)
#else
#define LOG2F(x) log2f(x)
#endif

__device__ __forceinline__ void wait_async0() {
#if __has_builtin(__builtin_amdgcn_s_wait_asynccnt)
  __builtin_amdgcn_s_wait_asynccnt(0);
#else
  asm volatile("s_wait_asynccnt 0" ::: "memory");
#endif
}

// A-matrix tile (16 rows x 32 K, bf16, row-major memory, 16B-aligned rows).
// Lane L (<16) holds row L, K chunks {k0+0..7, k0+16..23};
// lane L (>=16) holds row L-16, K chunks {k0+8..15, k0+24..31}.  (ISA 7.12.2)
__device__ __forceinline__ v16bf load_a_tile(const __bf16* tile, int stride,
                                             int k0, int lane) {
  int row = lane & 15;
  int c0  = k0 + ((lane & 16) ? 8 : 0);
  const __bf16* p = tile + row * stride + c0;
  V16U u;
  u.q[0] = *(const u32x4*)(p);        // global_load_b128 / ds_load_b128
  u.q[1] = *(const u32x4*)(p + 16);
  return u.v;
}

// B-matrix tile (32 K x 16 N, bf16): lane L holds column n0+(L&15),
// K chunks {k0+0..7, k0+16..23} (L<16) or {k0+8..15, k0+24..31} (L>=16).
__device__ __forceinline__ v16bf load_b_tile(const __bf16* base, int stride,
                                             int k0, int n0, int lane) {
  const __bf16* p = base + n0 + (lane & 15);
  int c0 = k0 + ((lane & 16) ? 8 : 0);
  V16U u;
#pragma unroll
  for (int s = 0; s < 8; ++s) {
    u.v[s]     = p[(c0 + s) * stride];
    u.v[s + 8] = p[(c0 + 16 + s) * stride];
  }
  return u.v;
}

__device__ __forceinline__ v8f wmma_bf16(v16bf a, v16bf b, v8f c) {
  return __builtin_amdgcn_wmma_f32_16x16x32_bf16(false, a, false, b,
                                                 (short)0, c, false, false);
}

// ---------------------------------------------------------------------------
// Kernel 0: fp32 -> bf16 conversion + transposes.
//   xb[N][D], xT[D][N], Wqb[HZ][D], WkT[D][HZ]
// ---------------------------------------------------------------------------
__global__ void convert_kernel(const float* __restrict__ x,
                               const float* __restrict__ Wq,
                               const float* __restrict__ Wk,
                               unsigned short* __restrict__ xb_,
                               unsigned short* __restrict__ xT_,
                               unsigned short* __restrict__ Wqb_,
                               unsigned short* __restrict__ WkT_) {
  __bf16* xb  = reinterpret_cast<__bf16*>(xb_);
  __bf16* xT  = reinterpret_cast<__bf16*>(xT_);
  __bf16* Wqb = reinterpret_cast<__bf16*>(Wqb_);
  __bf16* WkT = reinterpret_cast<__bf16*>(WkT_);
  int i = blockIdx.x * blockDim.x + threadIdx.x;   // 0 .. N*D-1 (== HZ*D-1)
  int r = i / D_DIM;          // token n, or flat head*z index hz
  int d = i % D_DIM;
  float xv = x[i];
  xb[i]               = (__bf16)xv;
  xT[d * N_TOK + r]   = (__bf16)xv;
  Wqb[i]              = (__bf16)Wq[i];
  WkT[d * HZ_DIM + r] = (__bf16)Wk[i];
}

// ---------------------------------------------------------------------------
// Kernel 1a: K projection.  Kb[h][n][z] = sum_d x[n,d] * Wk[hz,d]
// ---------------------------------------------------------------------------
__global__ __launch_bounds__(32)
void proj_k_kernel(const unsigned short* __restrict__ xb_,
                   const unsigned short* __restrict__ WkT_,
                   unsigned short* __restrict__ Kb_) {
  const __bf16* xb  = reinterpret_cast<const __bf16*>(xb_);
  const __bf16* WkT = reinterpret_cast<const __bf16*>(WkT_);
  __bf16* Kb        = reinterpret_cast<__bf16*>(Kb_);
  int lane = threadIdx.x;
  int n0   = blockIdx.x * 16;
  int hz0  = blockIdx.y * 16;
  v8f c = {};
#pragma unroll
  for (int dk = 0; dk < D_DIM / 32; ++dk) {
    v16bf a = load_a_tile(xb + n0 * D_DIM, D_DIM, dk * 32, lane);
    v16bf b = load_b_tile(WkT, HZ_DIM, dk * 32, hz0, lane);
    c = wmma_bf16(a, b, c);
  }
  // C layout: vgpr r -> M = r + (lane<16?0:8); N = lane&15
  int madd = (lane & 16) ? 8 : 0;
  int hz   = hz0 + (lane & 15);
  int h    = hz >> 6;
  int z    = hz & 63;
#pragma unroll
  for (int r = 0; r < 8; ++r) {
    int n = n0 + r + madd;
    Kb[((h * N_TOK + n) << 6) + z] = (__bf16)c[r];
  }
}

// ---------------------------------------------------------------------------
// Kernel 1b: Q projection (stored transposed).  Qt[hz][n] = sum_d Wq[hz,d]*x[n,d]
// ---------------------------------------------------------------------------
__global__ __launch_bounds__(32)
void proj_q_kernel(const unsigned short* __restrict__ Wqb_,
                   const unsigned short* __restrict__ xT_,
                   unsigned short* __restrict__ Qt_) {
  const __bf16* Wqb = reinterpret_cast<const __bf16*>(Wqb_);
  const __bf16* xT  = reinterpret_cast<const __bf16*>(xT_);
  __bf16* Qt        = reinterpret_cast<__bf16*>(Qt_);
  int lane = threadIdx.x;
  int hz0  = blockIdx.x * 16;
  int n0   = blockIdx.y * 16;
  v8f c = {};
#pragma unroll
  for (int dk = 0; dk < D_DIM / 32; ++dk) {
    v16bf a = load_a_tile(Wqb + hz0 * D_DIM, D_DIM, dk * 32, lane);
    v16bf b = load_b_tile(xT, N_TOK, dk * 32, n0, lane);
    c = wmma_bf16(a, b, c);
  }
  int madd = (lane & 16) ? 8 : 0;
  int n    = n0 + (lane & 15);
#pragma unroll
  for (int r = 0; r < 8; ++r) {
    int hz = hz0 + r + madd;
    Qt[hz * N_TOK + n] = (__bf16)c[r];
  }
}

// ---------------------------------------------------------------------------
// Kernel 2: one workgroup (8 waves) = one head x 8 query tiles.
// K panel streamed through LDS with double-buffered 64 KB chunks using the
// CDNA5 async global->LDS engine (ASYNCcnt); prefetch of chunk c+1 overlaps
// the WMMA + online-LSE compute on chunk c.  8x K-panel reuse per block.
// ---------------------------------------------------------------------------
#define KEYS_PER_CHUNK 512            // 512*64*2B = 64 KB per buffer
#define CHUNKS (N_TOK / KEYS_PER_CHUNK)
#define CHUNK_VEC ((KEYS_PER_CHUNK * Z_DIM) / 8)        // u32x4 elements: 4096
#define VEC_PER_THREAD (CHUNK_VEC / 256)                // 16

// All 256 threads cooperatively copy one 64 KB chunk (async when available).
__device__ __forceinline__ void stage_chunk(const __bf16* gsrc, __bf16* ldst,
                                            int tid) {
#if __has_builtin(__builtin_amdgcn_global_load_async_to_lds_b128)
  // Builtin signature (from hipcc diagnostic): arg0 = int4 addrspace(1)*,
  // arg1 = int4 addrspace(3)*, then two immediates (offset, cpol).
  typedef int b128_t __attribute__((vector_size(16)));
  typedef __attribute__((address_space(1))) b128_t* gp_t;
  typedef __attribute__((address_space(3))) b128_t* lp_t;
#pragma unroll
  for (int i = 0; i < VEC_PER_THREAD; ++i) {
    int idx = tid + i * 256;
    __builtin_amdgcn_global_load_async_to_lds_b128(
        (gp_t)((const u32x4*)gsrc + idx),
        (lp_t)((u32x4*)ldst + idx), 0, 0);
  }
#else
#pragma unroll 4
  for (int i = 0; i < VEC_PER_THREAD; ++i) {
    int idx = tid + i * 256;
    ((u32x4*)ldst)[idx] = ((const u32x4*)gsrc)[idx];
  }
#endif
}

__global__ __launch_bounds__(256)
void attn_lse_kernel(const unsigned short* __restrict__ Qt_,
                     const unsigned short* __restrict__ Kb_,
                     float* __restrict__ out) {
  const __bf16* Qt = reinterpret_cast<const __bf16*>(Qt_);
  const __bf16* Kb = reinterpret_cast<const __bf16*>(Kb_);

  __shared__ __bf16 ks[2][KEYS_PER_CHUNK * Z_DIM];   // 2 x 64 KB

  int tid  = threadIdx.x;
  int lane = tid & 31;
  int wid  = tid >> 5;                            // 0..7
  int h    = blockIdx.x >> 4;                     // 16 blocks per head
  int q0   = (((blockIdx.x & 15) << 3) + wid) << 4;

  const __bf16* Qh = Qt + (h * Z_DIM) * N_TOK;
  const __bf16* Kh = Kb + (h * N_TOK) * Z_DIM;

  // Prefetch chunk 0 while we pull the loop-invariant B operand (Q^T).
  stage_chunk(Kh, ks[0], tid);
  v16bf b0 = load_b_tile(Qh, N_TOK, 0,  q0, lane);
  v16bf b1 = load_b_tile(Qh, N_TOK, 32, q0, lane);
  wait_async0();
  __syncthreads();

  // exp2-domain online logsumexp state (per lane = per query column).
  const float beta2 = 0.12751744f;    // (1/sqrt(128)) * log2(e)
  float mrun = -INFINITY, srun = 0.0f;

  for (int c = 0; c < CHUNKS; ++c) {
    const __bf16* cur = ks[c & 1];
    if (c + 1 < CHUNKS)   // prefetch next chunk into the idle buffer
      stage_chunk(Kh + (c + 1) * KEYS_PER_CHUNK * Z_DIM, ks[(c + 1) & 1], tid);

    // ---- 16 iterations x 32 keys (two 16x16 score tiles per update) ----
    for (int kt = 0; kt < KEYS_PER_CHUNK / 32; ++kt) {
      const __bf16* At0 = cur + (kt * 32) * Z_DIM;
      const __bf16* At1 = At0 + 16 * Z_DIM;
      v16bf a00 = load_a_tile(At0, Z_DIM, 0,  lane);   // ds_load_b128
      v16bf a01 = load_a_tile(At0, Z_DIM, 32, lane);
      v16bf a10 = load_a_tile(At1, Z_DIM, 0,  lane);
      v16bf a11 = load_a_tile(At1, Z_DIM, 32, lane);
      v8f c0 = {}, c1 = {};
      c0 = wmma_bf16(a00, b0, c0);
      c0 = wmma_bf16(a01, b1, c0);
      c1 = wmma_bf16(a10, b0, c1);
      c1 = wmma_bf16(a11, b1, c1);

      // raw max over 16 key scores for this lane's query column
      float tm = fmaxf(c0[0], c1[0]);
#pragma unroll
      for (int r = 1; r < 8; ++r) tm = fmaxf(tm, fmaxf(c0[r], c1[r]));
      float newm = fmaxf(mrun, tm * beta2);
      float s = srun * EXP2F(mrun - newm);
#pragma unroll
      for (int r = 0; r < 8; ++r) {
        s += EXP2F(__builtin_fmaf(c0[r], beta2, -newm));
        s += EXP2F(__builtin_fmaf(c1[r], beta2, -newm));
      }
      mrun = newm;
      srun = s;
    }

    // My prefetch landed + everyone is done with `cur` before it is reused.
    wait_async0();
    __syncthreads();
  }

  // Lanes L and L^16 hold key-halves of the same query: merge (base-2).
  float m2 = __shfl_xor(mrun, 16, 32);
  float s2 = __shfl_xor(srun, 16, 32);
  float M  = fmaxf(mrun, m2);
  float S  = srun * EXP2F(mrun - M) + s2 * EXP2F(m2 - M);
  float A  = M + LOG2F(S);            // log2-sum-exp2 for this (h, q)

  // Sum over lanes (each query appears in both half-waves -> *0.5),
  // then scale by -ln(2)/beta to convert base-2 LSE to the reference value.
#pragma unroll
  for (int off = 16; off > 0; off >>= 1) A += __shfl_xor(A, off, 32);
  if (lane == 0)
    atomicAdd(out, A * (-3.9210315427464785f));   // 0.5 * -sqrt(128)*ln2
}

// ---------------------------------------------------------------------------
extern "C" void kernel_launch(void* const* d_in, const int* in_sizes, int n_in,
                              void* d_out, int out_size, void* d_ws, size_t ws_size,
                              hipStream_t stream) {
  const float* x  = (const float*)d_in[0];
  const float* Wq = (const float*)d_in[1];
  const float* Wk = (const float*)d_in[2];
  float* out = (float*)d_out;

  const size_t SMALL = (size_t)HZ_DIM * D_DIM * sizeof(unsigned short); // 512 KB
  const size_t BIG   = (size_t)H_HEADS * N_TOK * Z_DIM * sizeof(unsigned short); // 8 MB
  char* ws = (char*)d_ws;
  unsigned short* xb  = (unsigned short*)(ws + 0 * SMALL);
  unsigned short* xT  = (unsigned short*)(ws + 1 * SMALL);
  unsigned short* Wqb = (unsigned short*)(ws + 2 * SMALL);
  unsigned short* WkT = (unsigned short*)(ws + 3 * SMALL);
  unsigned short* Kb  = (unsigned short*)(ws + 4 * SMALL);
  unsigned short* Qt  = (unsigned short*)(ws + 4 * SMALL + BIG);

  (void)hipMemsetAsync(d_out, 0, sizeof(float), stream);

  convert_kernel<<<(N_TOK * D_DIM) / 256, 256, 0, stream>>>(
      x, Wq, Wk, xb, xT, Wqb, WkT);

  proj_k_kernel<<<dim3(N_TOK / 16, HZ_DIM / 16), 32, 0, stream>>>(xb, WkT, Kb);
  proj_q_kernel<<<dim3(HZ_DIM / 16, N_TOK / 16), 32, 0, stream>>>(Wqb, xT, Qt);

  // 32 heads * 16 blocks; each block = 8 waves = 8 query tiles of one head.
  attn_lse_kernel<<<H_HEADS * 16, 256, 0, stream>>>(Qt, Kb, out);
}
